// Time_att_47021301956871
// MI455X (gfx1250) — matrix operand order repaired and checked by reading, not compiled, not run on hardware
//
#include <hip/hip_runtime.h>
#include <hip/hip_bf16.h>

// Problem constants
#define Bb 16
#define Nn 128
#define Tt 96
#define Cc 256

typedef __attribute__((ext_vector_type(16))) __bf16 v16bf;
typedef __attribute__((ext_vector_type(8)))  float v8f;
typedef __attribute__((ext_vector_type(2)))  __bf16 v2bf;
typedef __attribute__((ext_vector_type(4)))  __bf16 v4bf;

union BFrag {
    uint4 q[2];
    v16bf v;
};

__device__ __forceinline__ v8f wmma_bf16(v16bf a, v16bf b, v8f c) {
    // v_wmma_f32_16x16x32_bf16: (neg_a, A, neg_b, B, c_mod, C, reuse_a, reuse_b)
    return __builtin_amdgcn_wmma_f32_16x16x32_bf16(false, a, false, b, (short)0, c,
                                                   false, false);
}

// ---- Pre-pass: convert Wq|Wk|Wv (f32, 256x256 each) to bf16 in workspace ----
// dst layout (bf16): [0,65536) Wq, [65536,131072) Wk, [131072,196608) Wv, row-major.
__global__ __launch_bounds__(256)
void convert_weights(const float* __restrict__ Wq, const float* __restrict__ Wk,
                     const float* __restrict__ Wv, unsigned* __restrict__ dst)
{
    int i = blockIdx.x * 256 + threadIdx.x;      // pair index, 3*32768 total
    const float* src = (i < 32768) ? Wq : (i < 65536) ? Wk : Wv;
    int j = i & 32767;
    float2 v2 = ((const float2*)src)[j];
    v2bf p = { (__bf16)v2.x, (__bf16)v2.y };
    dst[i] = __builtin_bit_cast(unsigned, p);
}

// One block per (b,n): fused proj(Q,K,V) -> attn softmax -> attn@V -> residual+LN.
// 512 threads = 16 wave32; wave w owns output-channel rows 16w..16w+15 throughout.
// Dynamic LDS (bf16 elems): X:[0,24576) as (T,C); Kt:[24576,49152) as (C,T);
// Vt:[49152,73728) as (T,C). X region reused as per-wave attn bounce after GEMM2.
__global__ __launch_bounds__(512)
void time_att_fused(const float* __restrict__ x,
                    const __bf16* __restrict__ Wbf,
                    const float* __restrict__ bq,
                    const float* __restrict__ bk,
                    const float* __restrict__ bv,
                    const float* __restrict__ w_out, const float* __restrict__ b_out,
                    const float* __restrict__ gamma, const float* __restrict__ beta,
                    float* __restrict__ out)
{
    extern __shared__ __align__(16) __bf16 smem[];
    const int KOFF = Tt * Cc;       // 24576
    const int VOFF = 2 * Tt * Cc;   // 49152

    const int tid  = threadIdx.x;
    const int w    = tid >> 5;      // wave id, 0..15
    const int lane = tid & 31;
    const int lm   = lane & 15;
    const int h    = lane >> 4;     // half-wave
    const int aoff = h ? 8 : 0;     // A-layout K sub-offset per half
    const int boff = h * 8;         // B-layout K sub-offset per half

    const int bn = blockIdx.x;
    const int bb = bn >> 7;
    const int nn = bn & 127;
    const size_t slab = (size_t)bn * (Tt * Cc);
    const float* xs = x + slab;

    // ---- Stage X slab (T,C) as bf16 into LDS (float4 -> 4x bf16 per step) ----
    {
        const float4* src = (const float4*)xs;
        unsigned long long* dst = (unsigned long long*)smem;
        for (int i = tid; i < (Tt * Cc / 4); i += 512) {
            float4 v4 = src[i];
            v4bf p = { (__bf16)v4.x, (__bf16)v4.y, (__bf16)v4.z, (__bf16)v4.w };
            dst[i] = __builtin_bit_cast(unsigned long long, p);
        }
    }
    __syncthreads();

    const v8f vzero = {0.f, 0.f, 0.f, 0.f, 0.f, 0.f, 0.f, 0.f};

    // ---- GEMM1 (fused Q,K,V): Out[o][t] = sum_c W[o][c] * X[t][c].
    // Each X B-frag is loaded from LDS ONCE and feeds three WMMAs (Wq/Wk/Wv A-frags),
    // cutting GEMM1 LDS read traffic 3x. Wave's M-tile = channel rows 16w..16w+15.
    v8f accQ[6], accK[6], accV[6];
    #pragma unroll
    for (int i = 0; i < 6; i++) { accQ[i] = vzero; accK[i] = vzero; accV[i] = vzero; }
    {
        const __bf16* wrq = Wbf + (16 * w + lm) * 256;
        const __bf16* wrk = wrq + 65536;
        const __bf16* wrv = wrq + 131072;
        #pragma unroll 1
        for (int kk = 0; kk < 8; kk++) {
            BFrag Aq, Ak, Av;
            Aq.q[0] = *(const uint4*)(wrq + kk * 32 + aoff);
            Aq.q[1] = *(const uint4*)(wrq + kk * 32 + aoff + 16);
            Ak.q[0] = *(const uint4*)(wrk + kk * 32 + aoff);
            Ak.q[1] = *(const uint4*)(wrk + kk * 32 + aoff + 16);
            Av.q[0] = *(const uint4*)(wrv + kk * 32 + aoff);
            Av.q[1] = *(const uint4*)(wrv + kk * 32 + aoff + 16);
            #pragma unroll
            for (int nt = 0; nt < 6; nt++) {
                BFrag Bf;
                const uint4* bp = (const uint4*)(smem + (nt * 16 + lm) * 256 + kk * 32 + boff);
                Bf.q[0] = bp[0]; Bf.q[1] = bp[2];
                accQ[nt] = wmma_bf16(Aq.v, Bf.v, accQ[nt]);
                accK[nt] = wmma_bf16(Ak.v, Bf.v, accK[nt]);
                accV[nt] = wmma_bf16(Av.v, Bf.v, accV[nt]);
            }
        }
    }

    // ---- Q: bias + bounce wave-private strip through Kt slice to get A-layout ----
    {
        __bf16* qs = smem + KOFF + w * (16 * Tt);
        #pragma unroll
        for (int r = 0; r < 8; r++) {
            float bias = bq[16 * w + 8 * h + r];
            #pragma unroll
            for (int nt = 0; nt < 6; nt++)
                qs[(8 * h + r) * Tt + nt * 16 + lm] = (__bf16)(accQ[nt][r] + bias);
        }
    }
    BFrag aQ[3];
    {
        const __bf16* qs = smem + KOFF + w * (16 * Tt);
        #pragma unroll
        for (int kk = 0; kk < 3; kk++) {   // K = T = 96 -> 3 steps of 32
            aQ[kk].q[0] = *(const uint4*)(qs + lm * Tt + kk * 32 + aoff);
            aQ[kk].q[1] = *(const uint4*)(qs + lm * Tt + kk * 32 + aoff + 16);
        }
    }

    // ---- K: bias + store Kt (C,T) (overwrites own Q bounce slice, safe in-order) ----
    #pragma unroll
    for (int r = 0; r < 8; r++) {
        int c = 16 * w + 8 * h + r;
        float bias = bk[c];
        #pragma unroll
        for (int nt = 0; nt < 6; nt++)
            smem[KOFF + c * Tt + nt * 16 + lm] = (__bf16)(accK[nt][r] + bias);
    }

    // ---- V: bias + store Vt (T,C); 4 consecutive channels pack into one b64 store ----
    #pragma unroll
    for (int g = 0; g < 2; g++) {
        int c0 = 16 * w + 8 * h + 4 * g;
        float b0v = bv[c0 + 0], b1v = bv[c0 + 1], b2v = bv[c0 + 2], b3v = bv[c0 + 3];
        #pragma unroll
        for (int nt = 0; nt < 6; nt++) {
            int t = nt * 16 + lm;
            v4bf p = { (__bf16)(accV[nt][4 * g + 0] + b0v),
                       (__bf16)(accV[nt][4 * g + 1] + b1v),
                       (__bf16)(accV[nt][4 * g + 2] + b2v),
                       (__bf16)(accV[nt][4 * g + 3] + b3v) };
            *(unsigned long long*)(smem + VOFF + t * 256 + c0) =
                __builtin_bit_cast(unsigned long long, p);
        }
    }
    __syncthreads();

    // ---- GEMM2: attn[i][j] = sum_t q[i][t] * k[t][j]; wave owns rows 16w..16w+15 ----
    v8f attn[16];
    #pragma unroll
    for (int i = 0; i < 16; i++) attn[i] = vzero;
    #pragma unroll
    for (int kk = 0; kk < 3; kk++) {
        #pragma unroll
        for (int jt = 0; jt < 16; jt++) {
            BFrag Bf;
            const uint4* bp = (const uint4*)(smem + KOFF + (jt * 16 + lm) * Tt + kk * 32 + boff);
            Bf.q[0] = bp[0]; Bf.q[1] = bp[2];
            attn[jt] = wmma_bf16(aQ[kk].v, Bf.v, attn[jt]);
        }
    }

    // ---- softmax over rows (length 256): per-lane over 16 tiles + half-wave reduction ----
    const float scale = 0.17677669529663687f;  // 1/sqrt(C/H) = 1/sqrt(32)
    #pragma unroll
    for (int r = 0; r < 8; r++) {
        float mx = -3.0e38f;
        #pragma unroll
        for (int jt = 0; jt < 16; jt++) mx = fmaxf(mx, attn[jt][r]);
        #pragma unroll
        for (int m = 8; m >= 1; m >>= 1) mx = fmaxf(mx, __shfl_xor(mx, m, 16));
        mx *= scale;
        float sum = 0.f;
        #pragma unroll
        for (int jt = 0; jt < 16; jt++) {
            float e = __expf(attn[jt][r] * scale - mx);
            attn[jt][r] = e;
            sum += e;
        }
        #pragma unroll
        for (int m = 8; m >= 1; m >>= 1) sum += __shfl_xor(sum, m, 16);
        float inv = 1.0f / sum;
        #pragma unroll
        for (int jt = 0; jt < 16; jt++) attn[jt][r] *= inv;
    }

    // ---- GEMM3: line = attn @ v; consume attn in 64-wide K chunks via per-wave LDS bounce ----
    v8f line[6];
    #pragma unroll
    for (int i = 0; i < 6; i++) line[i] = vzero;
    __bf16* bnc = smem + w * 1024;  // reuse X region: 16x64 bf16 per wave
    #pragma unroll
    for (int cc = 0; cc < 4; cc++) {
        #pragma unroll
        for (int j4 = 0; j4 < 4; j4++) {
            #pragma unroll
            for (int r = 0; r < 8; r++)
                bnc[(8 * h + r) * 64 + j4 * 16 + lm] = (__bf16)attn[cc * 4 + j4][r];
        }
        #pragma unroll
        for (int kk2 = 0; kk2 < 2; kk2++) {
            BFrag Af;
            Af.q[0] = *(const uint4*)(bnc + lm * 64 + kk2 * 32 + aoff);
            Af.q[1] = *(const uint4*)(bnc + lm * 64 + kk2 * 32 + aoff + 16);
            #pragma unroll
            for (int nt = 0; nt < 6; nt++) {
                BFrag Bf;
                const uint4* bp =
                    (const uint4*)(smem + VOFF + (nt * 16 + lm) * 256 + cc * 64 + kk2 * 32 + boff);
                Bf.q[0] = bp[0]; Bf.q[1] = bp[2];
                line[nt] = wmma_bf16(Af.v, Bf.v, line[nt]);
            }
        }
    }

    // ---- epilogue: head-combine + residual + LayerNorm over T, store (B,C,N,T) ----
    float sw = 0.f;
    #pragma unroll
    for (int i = 0; i < 8; i++) sw += w_out[i];
    const float b0 = b_out[0];

    #pragma unroll
    for (int nt = 0; nt < 6; nt++) {
        int t = nt * 16 + lm;
        #pragma unroll
        for (int r = 0; r < 8; r++) {
            int c = 16 * w + 8 * h + r;
            line[nt][r] = xs[t * 256 + c] + line[nt][r] * sw + b0;
        }
    }
    #pragma unroll
    for (int r = 0; r < 8; r++) {
        int c = 16 * w + 8 * h + r;
        float s = 0.f;
        #pragma unroll
        for (int nt = 0; nt < 6; nt++) s += line[nt][r];
        #pragma unroll
        for (int m = 8; m >= 1; m >>= 1) s += __shfl_xor(s, m, 16);
        float mu = s * (1.0f / 96.0f);
        float vs = 0.f;
        #pragma unroll
        for (int nt = 0; nt < 6; nt++) { float d = line[nt][r] - mu; vs += d * d; }
        #pragma unroll
        for (int m = 8; m >= 1; m >>= 1) vs += __shfl_xor(vs, m, 16);
        float inv = rsqrtf(vs * (1.0f / 96.0f) + 1e-5f);
        float* op = out + ((size_t)(bb * Cc + c) * Nn + nn) * Tt;
        #pragma unroll
        for (int nt = 0; nt < 6; nt++) {
            int t = nt * 16 + lm;
            op[t] = (line[nt][r] - mu) * inv * gamma[t] + beta[t];
        }
    }
}

extern "C" void kernel_launch(void* const* d_in, const int* in_sizes, int n_in,
                              void* d_out, int out_size, void* d_ws, size_t ws_size,
                              hipStream_t stream) {
    const float* x     = (const float*)d_in[0];
    const float* Wq    = (const float*)d_in[1];
    const float* bq    = (const float*)d_in[2];
    const float* Wk    = (const float*)d_in[3];
    const float* bk    = (const float*)d_in[4];
    const float* Wv    = (const float*)d_in[5];
    const float* bv    = (const float*)d_in[6];
    const float* w_out = (const float*)d_in[7];
    const float* b_out = (const float*)d_in[8];
    const float* gamma = (const float*)d_in[9];
    const float* beta  = (const float*)d_in[10];
    float* out = (float*)d_out;

    // Workspace: bf16 copies of Wq|Wk|Wv (3 * 65536 * 2B = 384 KB)
    unsigned* wpacked = (unsigned*)d_ws;
    hipLaunchKernelGGL(convert_weights, dim3(3 * 32768 / 256), dim3(256), 0, stream,
                       Wq, Wk, Wv, wpacked);

    const size_t lds_bytes = (size_t)(3 * Tt * Cc) * sizeof(__bf16);  // 144 KB
    dim3 grid(Bb * Nn);   // 2048 blocks, one per (b,n)
    dim3 block(512);      // 16 wave32
    hipLaunchKernelGGL(time_att_fused, grid, block, lds_bytes, stream,
                       x, (const __bf16*)d_ws, bq, bk, bv,
                       w_out, b_out, gamma, beta, out);
}
